// SupervisedGraphSage_85315230368144
// MI455X (gfx1250) — matrix operand order repaired and checked by reading, not compiled, not run on hardware
//
#include <hip/hip_runtime.h>
#include <hip/hip_bf16.h>
#include <stdint.h>

// ---------------------------------------------------------------------------
// Problem constants (from reference): B=4096, M=8192, F=256, HID=256, C=64, H=2
// ---------------------------------------------------------------------------
#define B_ROWS  4096
#define M_NEIGH 8192
#define F_DIM   256
#define HID_DIM 256
#define C_OUT   64
#define H_HEADS 2
#define KT_TILES (M_NEIGH / 32)    // 256 K-tiles in the attention GEMM

typedef __attribute__((ext_vector_type(16))) __bf16 v16bf;
typedef __attribute__((ext_vector_type(8)))  float  v8f;
typedef __attribute__((ext_vector_type(4)))  float  f32x4;

union AFrag { uint32_t u[8]; v16bf v; };
union BFrag { uint4 q[2];    v16bf v; };

__device__ __forceinline__ uint32_t pack_bf16x2(float a, float b) {
  // round-to-nearest-even f32 -> bf16, packed (a=low, b=high)
  uint32_t ua = __float_as_uint(a);
  uint32_t ub = __float_as_uint(b);
  ua += 0x7fffu + ((ua >> 16) & 1u);
  ub += 0x7fffu + ((ub >> 16) & 1u);
  return (ua >> 16) | (ub & 0xffff0000u);
}

// Issue one 16KB B-tile -> LDS via async DMA (gfx1250, tracked by ASYNCcnt).
// 128 threads x 8 chunks x 16B. g = tile base + tid*4 dwords,
// lds_byte_off = LDS byte offset of this thread's first chunk.
__device__ __forceinline__ void async_copy_tile(const uint32_t* __restrict__ g,
                                                uint32_t lds_byte_off) {
#pragma unroll
  for (int i = 0; i < 8; ++i) {
    const uint32_t* gp = g + i * 512;        // +128 chunks of 4 dwords
    uint32_t lo = lds_byte_off + i * 2048;   // +128 chunks of 16 bytes
    asm volatile("global_load_async_to_lds_b128 %0, %1, off"
                 :: "v"(lo), "v"(gp) : "memory");
  }
}

// ---------------------------------------------------------------------------
// Kernel 1: rank-1 projections  selfdot[h][b] = self_feats[b,:] . a[h,:F]
//                               neighdot[h][m] = neigh_feats[m,:] . a[h,F:]
// one wave (32 lanes) per output element
// ---------------------------------------------------------------------------
__global__ void dot_kernel(const float* __restrict__ self_feats,
                           const float* __restrict__ neigh_feats,
                           const float* __restrict__ a,
                           float* __restrict__ selfdot,
                           float* __restrict__ neighdot) {
  int gw   = (blockIdx.x * blockDim.x + threadIdx.x) >> 5;
  int lane = threadIdx.x & 31;
  const int per = B_ROWS + M_NEIGH;
  int h = gw / per;
  if (h >= H_HEADS) return;
  int idx = gw % per;

  const float* src;
  const float* av;
  float* dst;
  if (idx < B_ROWS) {
    src = self_feats + (size_t)idx * F_DIM;
    av  = a + (size_t)h * 2 * F_DIM;
    dst = selfdot + (size_t)h * B_ROWS + idx;
  } else {
    int c = idx - B_ROWS;
    src = neigh_feats + (size_t)c * F_DIM;
    av  = a + (size_t)h * 2 * F_DIM + F_DIM;
    dst = neighdot + (size_t)h * M_NEIGH + c;
  }
  float s = 0.f;
#pragma unroll
  for (int j = 0; j < F_DIM; j += 32) s += src[j + lane] * av[j + lane];
#pragma unroll
  for (int off = 16; off >= 1; off >>= 1) s += __shfl_xor(s, off, 32);
  if (lane == 0) *dst = s;
}

// ---------------------------------------------------------------------------
// Kernel 2: pre-pack a row-major f32 (K x N) matrix into WMMA bf16 B-fragment
// blocks. Block (kt,nt) covers K rows [kt*32, kt*32+32), N cols [nt*16, nt*16+16).
// Per ISA layout: lanes 0-15 hold K={0..7,16..23} (2 per VGPR), lanes 16-31 K={8..15,24..31},
// N = lane%16. Each lane's 8 dwords stored contiguously (lane*8) so the
// consumer does two b128 loads per lane.
// Launch: blockDim=32, gridDim = (K/32)*(N/16)
// ---------------------------------------------------------------------------
__global__ void convertB_kernel(const float* __restrict__ src, int N, int ntBlocks,
                                uint32_t* __restrict__ dst) {
  int b    = blockIdx.x;
  int lane = threadIdx.x & 31;
  int kt   = b / ntBlocks;
  int nt   = b % ntBlocks;
  int koff = kt * 32 + ((lane >> 4) << 3);   // lanes 0-15: +0, lanes 16-31: +8
  int n    = nt * 16 + (lane & 15);
  uint32_t* d = dst + (size_t)b * 256 + lane * 8;
#pragma unroll
  for (int i = 0; i < 8; ++i) {
    float f0 = src[(size_t)(koff + 2 * i + ((i >> 2) << 3))     * N + n];
    float f1 = src[(size_t)(koff + 2 * i + ((i >> 2) << 3) + 1) * N + n];
    d[i] = pack_bf16x2(f0, f1);
  }
}

// ---------------------------------------------------------------------------
// Kernel 3: fused flash-style masked-softmax attention + aggregation.
// One wave owns 16 rows x 256 cols of f32 accumulator (16 WMMA C tiles).
// B tiles (16KB each) are shared by the 4 waves of the block, so they are
// staged in LDS by async DMA (double-buffered): tile kt+1 streams in while
// scores/exp for tile kt are computed; waves then read fragments from LDS.
// mask is a 128MB single-pass stream -> non-temporal loads so it does not
// evict the L2-resident blocked neigh matrix.
// Softmax max-shift omitted: scores are O(1) by construction (a ~ 1/sqrt(2F)),
// softmax is shift-invariant. Row sum accumulated, divided out at the end.
// grid = (B/64, H), block = 128 (4 waves, 16 rows each, identical columns)
// ---------------------------------------------------------------------------
__global__ __launch_bounds__(128) void flash_attn_agg(
    const float* __restrict__ mask,        // B x M
    const float* __restrict__ selfdot,     // H x B
    const float* __restrict__ neighdot,    // H x M
    const uint32_t* __restrict__ nbB,      // (M/32)*(F/16) blocks x 256 dwords
    float* __restrict__ agg) {             // H x B x F
  __shared__ uint32_t bS[2][4096];         // 2 x 16KB B-tile double buffer

  int h    = blockIdx.y;
  int tid  = threadIdx.x;
  int lane = tid & 31;
  int wave = tid >> 5;
  int hi   = lane >> 4;                    // 0 for lanes 0-15, 1 for 16-31
  int r_base = blockIdx.x * 64 + wave * 16;
  int row    = r_base + (lane & 15);

  float ds = selfdot[(size_t)h * B_ROWS + row];
  const float* nd   = neighdot + (size_t)h * M_NEIGH;
  const float* mrow = mask + (size_t)row * M_NEIGH;

  v8f acc[16];
  const v8f zero8 = {0.f, 0.f, 0.f, 0.f, 0.f, 0.f, 0.f, 0.f};
#pragma unroll
  for (int t = 0; t < 16; ++t) acc[t] = zero8;

  float psum = 0.f;

  // prologue: start DMA of tile 0 into buffer 0
  async_copy_tile(nbB + tid * 4, (uint32_t)(size_t)&bS[0][tid * 4]);

  for (int kt = 0; kt < KT_TILES; ++kt) {
    int buf = kt & 1;
    if (kt + 1 < KT_TILES) {
      async_copy_tile(nbB + (size_t)(kt + 1) * 4096 + tid * 4,
                      (uint32_t)(size_t)&bS[buf ^ 1][tid * 4]);
    }

    // ---- build P tile (overlaps the async DMA) ----
    // A-fragment column layout: lanes 0-15 hold K={0..7,16..23},
    // lanes 16-31 hold K={8..15,24..31} of this 32-wide tile.
    int c0 = kt * 32 + hi * 8;
    f32x4 mv[4], nv[4];
    mv[0] = __builtin_nontemporal_load((const f32x4*)(mrow + c0));
    mv[1] = __builtin_nontemporal_load((const f32x4*)(mrow + c0 + 4));
    mv[2] = __builtin_nontemporal_load((const f32x4*)(mrow + c0 + 16));
    mv[3] = __builtin_nontemporal_load((const f32x4*)(mrow + c0 + 20));
    nv[0] = *(const f32x4*)(nd + c0);
    nv[1] = *(const f32x4*)(nd + c0 + 4);
    nv[2] = *(const f32x4*)(nd + c0 + 16);
    nv[3] = *(const f32x4*)(nd + c0 + 20);
    const float* mm = (const float*)mv;
    const float* nn = (const float*)nv;

    float p[16];
#pragma unroll
    for (int i = 0; i < 16; ++i) {
      float s = ds + nn[i];
      s = (s > 0.f) ? s : 0.2f * s;                 // leaky_relu(0.2)
      float pv = (mm[i] > 0.f) ? __expf(s) : 0.f;   // masked exp
      p[i] = pv;
      psum += pv;
    }

    AFrag A;
#pragma unroll
    for (int i = 0; i < 8; ++i) A.u[i] = pack_bf16x2(p[2 * i], p[2 * i + 1]);

    // ---- wait for tile kt's DMA (async loads complete in order, so
    // "<= 8 outstanding" leaves only tile kt+1's transfers in flight) ----
    if (kt + 1 < KT_TILES)
      asm volatile("s_wait_asynccnt 0x8" ::: "memory");
    else
      asm volatile("s_wait_asynccnt 0x0" ::: "memory");
    __syncthreads();

    const uint32_t* bb = &bS[buf][lane * 8];
#pragma unroll
    for (int nt = 0; nt < 16; ++nt) {
      BFrag Bf;
      Bf.q[0] = *(const uint4*)(bb + nt * 256);
      Bf.q[1] = *(const uint4*)(bb + nt * 256 + 4);
      acc[nt] = __builtin_amdgcn_wmma_f32_16x16x32_bf16(
          false, A.v, false, Bf.v, (short)0, acc[nt], false, false);
    }
    // all waves done reading buf before next iteration's DMA overwrites it
    __syncthreads();
  }

  // per-row softmax denominator: lane l and lane l^16 hold complementary cols
  float rsum = psum + __shfl_xor(psum, 16, 32);
  float linv = 1.0f / rsum;   // mask[:,0]==1 guarantees rsum > 0

  // C layout: VGPR j -> rows j (lanes 0-15) / j+8 (lanes 16-31), col = lane%16
  float* aout = agg + ((size_t)h * B_ROWS + r_base) * F_DIM;
#pragma unroll
  for (int j = 0; j < 8; ++j) {
    int rr = j + hi * 8;
    float lv = __shfl(linv, rr, 32);   // row stats live in lanes 0..15
#pragma unroll
    for (int nt = 0; nt < 16; ++nt) {
      aout[(size_t)rr * F_DIM + nt * 16 + (lane & 15)] = acc[nt][j] * lv;
    }
  }
}

// ---------------------------------------------------------------------------
// Kernel 4: enc_h = relu([self_feats | agg_h] @ enc_w[h])   (4096x512 @ 512x256)
// grid = (B/64, H), block = 128; wave: 16 rows x 256 cols, K = 16 tiles of 32
// ---------------------------------------------------------------------------
__global__ __launch_bounds__(128) void enc_gemm(
    const float* __restrict__ self_feats,  // B x F
    const float* __restrict__ agg,         // H x B x F
    const uint32_t* __restrict__ encB,     // H x 256 blocks x 256 dwords
    float* __restrict__ enc) {             // B x (H*HID)
  int h    = blockIdx.y;
  int lane = threadIdx.x & 31;
  int wave = threadIdx.x >> 5;
  int hi   = lane >> 4;
  int r_base = blockIdx.x * 64 + wave * 16;
  int row    = r_base + (lane & 15);

  v8f acc[16];
  const v8f zero8 = {0.f, 0.f, 0.f, 0.f, 0.f, 0.f, 0.f, 0.f};
#pragma unroll
  for (int t = 0; t < 16; ++t) acc[t] = zero8;

  const uint32_t* bb = encB + (size_t)h * 65536 + lane * 8;

#pragma unroll
  for (int kt = 0; kt < 16; ++kt) {
    const float* src = (kt < 8)
        ? (self_feats + (size_t)row * F_DIM + kt * 32)
        : (agg + ((size_t)h * B_ROWS + row) * F_DIM + (kt - 8) * 32);
    int c0 = hi * 8;
    f32x4 xv[4];
    xv[0] = *(const f32x4*)(src + c0);
    xv[1] = *(const f32x4*)(src + c0 + 4);
    xv[2] = *(const f32x4*)(src + c0 + 16);
    xv[3] = *(const f32x4*)(src + c0 + 20);
    const float* xx = (const float*)xv;

    AFrag A;
#pragma unroll
    for (int i = 0; i < 8; ++i) A.u[i] = pack_bf16x2(xx[2 * i], xx[2 * i + 1]);

#pragma unroll
    for (int nt = 0; nt < 16; ++nt) {
      BFrag Bf;
      const uint32_t* bp = bb + (size_t)(kt * 16 + nt) * 256;
      Bf.q[0] = *(const uint4*)(bp);
      Bf.q[1] = *(const uint4*)(bp + 4);
      acc[nt] = __builtin_amdgcn_wmma_f32_16x16x32_bf16(
          false, A.v, false, Bf.v, (short)0, acc[nt], false, false);
    }
  }

  float* eout = enc + (size_t)r_base * (H_HEADS * HID_DIM) + (size_t)h * HID_DIM;
#pragma unroll
  for (int j = 0; j < 8; ++j) {
    int rr = j + hi * 8;
#pragma unroll
    for (int nt = 0; nt < 16; ++nt) {
      float v = acc[nt][j];
      v = (v > 0.f) ? v : 0.f;                       // relu
      eout[(size_t)rr * (H_HEADS * HID_DIM) + nt * 16 + (lane & 15)] = v;
    }
  }
}

// ---------------------------------------------------------------------------
// Kernel 5: out = enc @ out_w   (4096x512 @ 512x64)
// grid = B/64, block = 128; wave: 16 rows x 64 cols
// ---------------------------------------------------------------------------
__global__ __launch_bounds__(128) void out_gemm(
    const float* __restrict__ enc,         // B x 512
    const uint32_t* __restrict__ outB,     // 64 blocks x 256 dwords
    float* __restrict__ out) {             // B x C
  const int K = H_HEADS * HID_DIM;         // 512
  int lane = threadIdx.x & 31;
  int wave = threadIdx.x >> 5;
  int hi   = lane >> 4;
  int r_base = blockIdx.x * 64 + wave * 16;
  int row    = r_base + (lane & 15);

  v8f acc[4];
  const v8f zero8 = {0.f, 0.f, 0.f, 0.f, 0.f, 0.f, 0.f, 0.f};
#pragma unroll
  for (int t = 0; t < 4; ++t) acc[t] = zero8;

#pragma unroll
  for (int kt = 0; kt < 16; ++kt) {
    const float* src = enc + (size_t)row * K + kt * 32;
    int c0 = hi * 8;
    f32x4 xv[4];
    xv[0] = *(const f32x4*)(src + c0);
    xv[1] = *(const f32x4*)(src + c0 + 4);
    xv[2] = *(const f32x4*)(src + c0 + 16);
    xv[3] = *(const f32x4*)(src + c0 + 20);
    const float* xx = (const float*)xv;

    AFrag A;
#pragma unroll
    for (int i = 0; i < 8; ++i) A.u[i] = pack_bf16x2(xx[2 * i], xx[2 * i + 1]);

#pragma unroll
    for (int nt = 0; nt < 4; ++nt) {
      BFrag Bf;
      const uint32_t* bp = outB + (size_t)(kt * 4 + nt) * 256 + lane * 8;
      Bf.q[0] = *(const uint4*)(bp);
      Bf.q[1] = *(const uint4*)(bp + 4);
      acc[nt] = __builtin_amdgcn_wmma_f32_16x16x32_bf16(
          false, A.v, false, Bf.v, (short)0, acc[nt], false, false);
    }
  }

#pragma unroll
  for (int j = 0; j < 8; ++j) {
    int rr = j + hi * 8;
#pragma unroll
    for (int nt = 0; nt < 4; ++nt) {
      out[(size_t)(r_base + rr) * C_OUT + nt * 16 + (lane & 15)] = acc[nt][j];
    }
  }
}

// ---------------------------------------------------------------------------
// Host-side orchestration. Workspace layout (units: floats/dwords):
//   selfdot  : [0, 8192)                     H*B
//   neighdot : [8192, 24576)                 H*M
//   agg      : [24576, 2121728)              H*B*F   f32
//   enc      : [2121728, 4218880)            B*512   f32
//   nbB      : [4218880, 5267456)            (M/32)*(F/16)*256 dwords
//   encB     : [5267456, 5398528)            H*16*16*256 dwords
//   outB     : [5398528, 5414912)            16*4*256 dwords
// total ~21.7 MB
// ---------------------------------------------------------------------------
extern "C" void kernel_launch(void* const* d_in, const int* in_sizes, int n_in,
                              void* d_out, int out_size, void* d_ws, size_t ws_size,
                              hipStream_t stream) {
  const float* self_feats  = (const float*)d_in[0];
  const float* neigh_feats = (const float*)d_in[1];
  const float* mask        = (const float*)d_in[2];
  const float* a           = (const float*)d_in[3];
  const float* enc_w       = (const float*)d_in[4];
  const float* out_w       = (const float*)d_in[5];
  float* out = (float*)d_out;

  float* wsf = (float*)d_ws;
  float*    selfdot  = wsf;
  float*    neighdot = wsf + 8192;
  float*    agg      = wsf + 24576;
  float*    enc      = wsf + 2121728;
  uint32_t* nbB      = (uint32_t*)(wsf + 4218880);
  uint32_t* encB     = (uint32_t*)(wsf + 5267456);
  uint32_t* outB     = (uint32_t*)(wsf + 5398528);

  // 1) rank-1 projections: H*(B+M) = 24576 waves, 8 waves per 256-thread block
  dot_kernel<<<3072, 256, 0, stream>>>(self_feats, neigh_feats, a, selfdot, neighdot);

  // 2) pre-pack all B operands into WMMA bf16 fragment blocks
  convertB_kernel<<<(M_NEIGH / 32) * (F_DIM / 16), 32, 0, stream>>>(
      neigh_feats, F_DIM, F_DIM / 16, nbB);                         // 4096 blocks
  convertB_kernel<<<16 * 16, 32, 0, stream>>>(
      enc_w, HID_DIM, HID_DIM / 16, encB);                          // head 0
  convertB_kernel<<<16 * 16, 32, 0, stream>>>(
      enc_w + (size_t)2 * F_DIM * HID_DIM, HID_DIM, HID_DIM / 16, encB + 65536); // head 1
  convertB_kernel<<<16 * 4, 32, 0, stream>>>(
      out_w, C_OUT, C_OUT / 16, outB);

  // 3) fused masked-softmax attention + aggregation (single pass over mask,
  //    async-to-LDS double-buffered B tiles, non-temporal mask stream)
  flash_attn_agg<<<dim3(B_ROWS / 64, H_HEADS), 128, 0, stream>>>(
      mask, selfdot, neighdot, nbB, agg);

  // 4) encoder GEMM + relu
  enc_gemm<<<dim3(B_ROWS / 64, H_HEADS), 128, 0, stream>>>(
      self_feats, agg, encB, enc);

  // 5) output GEMM
  out_gemm<<<B_ROWS / 64, 128, 0, stream>>>(enc, outB, out);
}
